// SimpleGNNModel_15788299780297
// MI455X (gfx1250) — matrix operand order repaired and checked by reading, not compiled, not run on hardware
//
#include <hip/hip_runtime.h>
#include <hip/hip_bf16.h>

typedef _Float16 v16h __attribute__((ext_vector_type(16)));
typedef _Float16 v8h  __attribute__((ext_vector_type(8)));
typedef float    v8f  __attribute__((ext_vector_type(8)));

#define N_NODES  100000
#define E_EDGES  1600000
#define IN_DIM   5
#define HID      64
#define EDGE_DIM 4
#define OUT_DIM  3
#define NLAYER   3
#define BN_EPS   1e-5f
#define KPAD     96   // HID + EDGE_DIM = 68, padded to 3 x 32
#define MT       4    // M-subtiles (of 16 edges) per wave -> 64 edges/wave
#define WAVES    4    // waves per block (128 threads)
#define EDGES_PER_BLOCK (WAVES * MT * 16)   // 256

// ---------------------------------------------------------------------------
// Weight prep: pad W1 [L,68,64] -> f16 [L,96,64] (zeros in pad), W2 -> f16.
// ---------------------------------------------------------------------------
__global__ void prep_weights(const float* __restrict__ W1,
                             const float* __restrict__ W2,
                             _Float16* __restrict__ W1p,
                             _Float16* __restrict__ W2h) {
  int i = blockIdx.x * blockDim.x + threadIdx.x;
  const int tot1 = NLAYER * KPAD * HID;
  if (i < tot1) {
    int l = i / (KPAD * HID);
    int r = i % (KPAD * HID);
    int k = r / HID, n = r % HID;
    float v = (k < HID + EDGE_DIM) ? W1[(l * (HID + EDGE_DIM) + k) * HID + n] : 0.0f;
    W1p[i] = (_Float16)v;
  }
  const int tot2 = NLAYER * HID * HID;
  if (i < tot2) W2h[i] = (_Float16)W2[i];
}

// ---------------------------------------------------------------------------
// Input MLP: h = relu(x @ Win + bin); keep f32 (residual) + f16 (WMMA gather)
// ---------------------------------------------------------------------------
__global__ void input_mlp(const float* __restrict__ x,
                          const float* __restrict__ Win,
                          const float* __restrict__ bin,
                          float* __restrict__ h,
                          _Float16* __restrict__ hh) {
  long i = (long)blockIdx.x * blockDim.x + threadIdx.x;
  if (i >= (long)N_NODES * HID) return;
  int n = (int)(i / HID), c = (int)(i % HID);
  float acc = bin[c];
#pragma unroll
  for (int k = 0; k < IN_DIM; k++) acc += x[n * IN_DIM + k] * Win[k * HID + c];
  acc = fmaxf(acc, 0.0f);
  h[i] = acc;
  hh[i] = (_Float16)acc;
}

// ---------------------------------------------------------------------------
// Edge MLP + scatter-add. One wave == 64 edges (4 subtiles of M=16), so each
// B fragment pulled from LDS is reused by 4 WMMAs.
//   GEMM1: [64 x 96(f16)] @ [96 x 64] -> relu -> (LDS transpose)
//   GEMM2: [64 x 64(f16)] @ [64 x 64] -> + b2 -> atomicAdd into agg[dst]
// ---------------------------------------------------------------------------
__global__ __launch_bounds__(128)
void edge_mlp_scatter(const int* __restrict__ src, const int* __restrict__ dst,
                      const float* __restrict__ ea, const _Float16* __restrict__ hh,
                      const _Float16* __restrict__ W1p, const float* __restrict__ b1,
                      const _Float16* __restrict__ W2h, const float* __restrict__ b2,
                      float* __restrict__ agg) {
  __shared__ __align__(32) _Float16 sW1[KPAD * HID];        // 12 KB
  __shared__ __align__(32) _Float16 sW2[HID * HID];         //  8 KB
  __shared__ float sb1[HID], sb2[HID];
  __shared__ __align__(16) _Float16 m1[WAVES][MT * 16 * HID]; // 32 KB transpose buf

  const int tid = threadIdx.x;
  for (int i = tid; i < KPAD * HID; i += 128) sW1[i] = W1p[i];
  for (int i = tid; i < HID * HID; i += 128) sW2[i] = W2h[i];
  if (tid < HID) { sb1[tid] = b1[tid]; sb2[tid] = b2[tid]; }
  __syncthreads();

  const int wave = tid >> 5, lane = tid & 31;
  const int e0 = (blockIdx.x * WAVES + wave) * (MT * 16);  // E = 6250*256 exactly
  const int row = lane & 15;                 // A-fragment: lane -> M
  const int khalf = (lane < 16) ? 0 : 8;     // K sub-offset per half-wave
  const int mbase = (lane < 16) ? 0 : 8;     // C/D-fragment: vgpr -> M

  union AF { v16h v; v8h h[2]; };

  // ---- A fragments (resident): gathered h[src] (K 0..63) + edge_attr pad ----
  AF a[MT][3];
#pragma unroll
  for (int m = 0; m < MT; m++) {
    const int e = e0 + m * 16 + row;
    const long srow = (long)src[e];
    const _Float16* hrow = hh + srow * HID;
    a[m][0].h[0] = *(const v8h*)(hrow + khalf);
    a[m][0].h[1] = *(const v8h*)(hrow + khalf + 16);
    a[m][1].h[0] = *(const v8h*)(hrow + 32 + khalf);
    a[m][1].h[1] = *(const v8h*)(hrow + 32 + khalf + 16);
#pragma unroll
    for (int i = 0; i < 8; i++) { a[m][2].h[0][i] = (_Float16)0.0f; a[m][2].h[1][i] = (_Float16)0.0f; }
    if (lane < 16) {
      const float* erow = ea + (long)e * EDGE_DIM;
#pragma unroll
      for (int i = 0; i < EDGE_DIM; i++) a[m][2].h[0][i] = (_Float16)erow[i];
    }
  }

  // ---- GEMM1: per N-tile load 3 B frags once, fire 4x3 WMMAs ----
#pragma unroll
  for (int t = 0; t < 4; t++) {
    v16h bk0, bk1, bk2;
#pragma unroll
    for (int s = 0; s < 16; s++) {
      const int kk = khalf + (s & 7) + ((s >> 3) << 4);
      const int col = t * 16 + (lane & 15);
      bk0[s] = sW1[( 0 + kk) * HID + col];
      bk1[s] = sW1[(32 + kk) * HID + col];
      bk2[s] = sW1[(64 + kk) * HID + col];
    }
    const int nloc = t * 16 + (lane & 15);
    const float bias = sb1[nloc];
#pragma unroll
    for (int m = 0; m < MT; m++) {
      v8f c = {};
      c = __builtin_amdgcn_wmma_f32_16x16x32_f16(false, a[m][0].v, false, bk0, (short)0, c, false, false);
      c = __builtin_amdgcn_wmma_f32_16x16x32_f16(false, a[m][1].v, false, bk1, (short)0, c, false, false);
      c = __builtin_amdgcn_wmma_f32_16x16x32_f16(false, a[m][2].v, false, bk2, (short)0, c, false, false);
#pragma unroll
      for (int v = 0; v < 8; v++) {
        float val = fmaxf(c[v] + bias, 0.0f);
        m1[wave][(m * 16 + mbase + v) * HID + nloc] = (_Float16)val;
      }
    }
  }
  __syncthreads();

  // ---- GEMM2 A fragments from LDS (row-major m1), resident ----
  AF p[MT][2];
#pragma unroll
  for (int m = 0; m < MT; m++) {
    const _Float16* mrow = &m1[wave][(m * 16 + row) * HID];
    p[m][0].h[0] = *(const v8h*)(mrow + khalf);
    p[m][0].h[1] = *(const v8h*)(mrow + khalf + 16);
    p[m][1].h[0] = *(const v8h*)(mrow + 32 + khalf);
    p[m][1].h[1] = *(const v8h*)(mrow + 32 + khalf + 16);
  }

  // dst indices for all 4 subtiles (8 rows per lane each)
  int dsti[MT][8];
#pragma unroll
  for (int m = 0; m < MT; m++)
#pragma unroll
    for (int v = 0; v < 8; v++) dsti[m][v] = dst[e0 + m * 16 + mbase + v];

  // ---- GEMM2: per N-tile load 2 B frags once, fire 4x2 WMMAs; scatter ----
#pragma unroll
  for (int t = 0; t < 4; t++) {
    v16h qk0, qk1;
#pragma unroll
    for (int s = 0; s < 16; s++) {
      const int kk = khalf + (s & 7) + ((s >> 3) << 4);
      const int col = t * 16 + (lane & 15);
      qk0[s] = sW2[( 0 + kk) * HID + col];
      qk1[s] = sW2[(32 + kk) * HID + col];
    }
    const int nloc = t * 16 + (lane & 15);
    const float bias = sb2[nloc];
#pragma unroll
    for (int m = 0; m < MT; m++) {
      v8f d = {};
      d = __builtin_amdgcn_wmma_f32_16x16x32_f16(false, p[m][0].v, false, qk0, (short)0, d, false, false);
      d = __builtin_amdgcn_wmma_f32_16x16x32_f16(false, p[m][1].v, false, qk1, (short)0, d, false, false);
#pragma unroll
      for (int v = 0; v < 8; v++) {
        atomicAdd(&agg[(long)dsti[m][v] * HID + nloc], d[v] + bias);
      }
    }
  }
}

// ---------------------------------------------------------------------------
// BN stats: per-channel sum / sumsq over nodes (block partials -> atomics)
// ---------------------------------------------------------------------------
__global__ void bn_stats(const float* __restrict__ agg, float* __restrict__ stats) {
  __shared__ float ssum[256], ssq[256];
  const int c = threadIdx.x & 63;
  const int g = threadIdx.x >> 6;   // 0..3 row groups
  float s = 0.0f, q = 0.0f;
  for (int r = g; r < 256; r += 4) {
    long node = (long)blockIdx.x * 256 + r;
    if (node < N_NODES) {
      float v = agg[node * HID + c];
      s += v; q += v * v;
    }
  }
  ssum[threadIdx.x] = s; ssq[threadIdx.x] = q;
  __syncthreads();
  if (threadIdx.x < 64) {
    float ts = ssum[threadIdx.x] + ssum[threadIdx.x + 64] + ssum[threadIdx.x + 128] + ssum[threadIdx.x + 192];
    float tq = ssq[threadIdx.x]  + ssq[threadIdx.x + 64]  + ssq[threadIdx.x + 128]  + ssq[threadIdx.x + 192];
    atomicAdd(&stats[threadIdx.x], ts);
    atomicAdd(&stats[64 + threadIdx.x], tq);
  }
}

// ---------------------------------------------------------------------------
// BN apply + relu + residual; refresh f32 and f16 copies of h
// ---------------------------------------------------------------------------
__global__ void bn_apply(const float* __restrict__ agg, const float* __restrict__ stats,
                         const float* __restrict__ gamma, const float* __restrict__ beta,
                         float* __restrict__ h, _Float16* __restrict__ hh) {
  long i = (long)blockIdx.x * blockDim.x + threadIdx.x;
  if (i >= (long)N_NODES * HID) return;
  const int c = (int)(i & 63);
  const float inv_n = 1.0f / (float)N_NODES;
  float mu  = stats[c] * inv_n;
  float var = stats[64 + c] * inv_n - mu * mu;
  float inv = rsqrtf(var + BN_EPS);
  float v = (agg[i] - mu) * inv * gamma[c] + beta[c];
  float nh = h[i] + fmaxf(v, 0.0f);
  h[i] = nh;
  hh[i] = (_Float16)nh;
}

// ---------------------------------------------------------------------------
// Output MLP: out = relu(h@W3+b3)@W4+b4; one block (64 threads) per node
// ---------------------------------------------------------------------------
__global__ void out_mlp(const float* __restrict__ h, const float* __restrict__ W3,
                        const float* __restrict__ b3, const float* __restrict__ W4,
                        const float* __restrict__ b4, float* __restrict__ out) {
  __shared__ float sh[HID];
  __shared__ float part[HID * OUT_DIM];
  const int n = blockIdx.x;
  const int j = threadIdx.x;
  sh[j] = h[(long)n * HID + j];
  __syncthreads();
  float t = b3[j];
#pragma unroll
  for (int k = 0; k < HID; k++) t += sh[k] * W3[k * HID + j];
  t = fmaxf(t, 0.0f);
#pragma unroll
  for (int o = 0; o < OUT_DIM; o++) part[j * OUT_DIM + o] = t * W4[j * OUT_DIM + o];
  __syncthreads();
  if (j < OUT_DIM) {
    float s = b4[j];
#pragma unroll
    for (int k = 0; k < HID; k++) s += part[k * OUT_DIM + j];
    out[n * OUT_DIM + j] = s;
  }
}

// ---------------------------------------------------------------------------
extern "C" void kernel_launch(void* const* d_in, const int* in_sizes, int n_in,
                              void* d_out, int out_size, void* d_ws, size_t ws_size,
                              hipStream_t stream) {
  const float* x     = (const float*)d_in[0];
  const int*   eidx  = (const int*)  d_in[1];   // [2, E]
  const float* ea    = (const float*)d_in[2];
  const float* Win   = (const float*)d_in[3];
  const float* bin   = (const float*)d_in[4];
  const float* W1    = (const float*)d_in[5];   // [L, 68, 64]
  const float* b1    = (const float*)d_in[6];   // [L, 64]
  const float* W2    = (const float*)d_in[7];   // [L, 64, 64]
  const float* b2    = (const float*)d_in[8];
  const float* gamma = (const float*)d_in[9];
  const float* beta  = (const float*)d_in[10];
  const float* W3    = (const float*)d_in[11];
  const float* b3    = (const float*)d_in[12];
  const float* W4    = (const float*)d_in[13];
  const float* b4    = (const float*)d_in[14];
  float* out = (float*)d_out;

  const int* src = eidx;
  const int* dst = eidx + E_EDGES;

  // Workspace carve-up (aligned offsets)
  char* ws = (char*)d_ws;
  float*    h    = (float*)   (ws + 0);          // 25,600,000 B
  _Float16* hh   = (_Float16*)(ws + 25600000);   // 12,800,000 B
  float*    agg  = (float*)   (ws + 38400000);   // 25,600,000 B
  _Float16* W1p  = (_Float16*)(ws + 64000000);   //     36,864 B
  _Float16* W2h  = (_Float16*)(ws + 64036864);   //     24,576 B
  float*    stats= (float*)   (ws + 64061440);   //        512 B

  // weights -> f16 (padded)
  {
    int tot = NLAYER * KPAD * HID;
    prep_weights<<<(tot + 255) / 256, 256, 0, stream>>>(W1, W2, W1p, W2h);
  }
  // input MLP
  {
    long tot = (long)N_NODES * HID;
    input_mlp<<<(int)((tot + 255) / 256), 256, 0, stream>>>(x, Win, bin, h, hh);
  }

  for (int l = 0; l < NLAYER; l++) {
    hipMemsetAsync(agg, 0, (size_t)N_NODES * HID * sizeof(float), stream);
    hipMemsetAsync(stats, 0, 2 * HID * sizeof(float), stream);

    edge_mlp_scatter<<<E_EDGES / EDGES_PER_BLOCK, 128, 0, stream>>>(
        src, dst, ea, hh,
        W1p + (size_t)l * KPAD * HID, b1 + l * HID,
        W2h + (size_t)l * HID * HID,  b2 + l * HID, agg);

    bn_stats<<<(N_NODES + 255) / 256, 256, 0, stream>>>(agg, stats);

    long tot = (long)N_NODES * HID;
    bn_apply<<<(int)((tot + 255) / 256), 256, 0, stream>>>(
        agg, stats, gamma + l * HID, beta + l * HID, h, hh);
  }

  out_mlp<<<N_NODES, HID, 0, stream>>>(h, W3, b3, W4, b4, out);
}
